// PCMDeeplab_68247030333703
// MI455X (gfx1250) — matrix operand order repaired and checked by reading, not compiled, not run on hardware
//
#include <hip/hip_runtime.h>

// ---------------------------------------------------------------------------
// Types for CDNA5 WMMA (f16 flavor: operands bounded in [-4,4] here, so f16's
// 10-bit mantissa beats bf16, and v_cvt_pk_f16_f32 packs 2 values in 1 VALU op)
// ---------------------------------------------------------------------------
typedef __attribute__((ext_vector_type(16))) _Float16 v16h;
typedef __attribute__((ext_vector_type(8)))  float    v8f;

struct H16Bits { uint4 a, b; };

// float -> f16 (RTZ, single v_cvt_pk_f16_f32)
__device__ __forceinline__ unsigned short f2h(float x) {
    return (unsigned short)__builtin_bit_cast(unsigned int,
               __builtin_amdgcn_cvt_pkrtz(x, 0.0f));
}

// relu + pack two floats into one dword of f16 pairs (one packed cvt op).
__device__ __forceinline__ unsigned int pack2_relu(float a, float b) {
    a = (a > 0.0f) ? a : 0.0f;
    b = (b > 0.0f) ? b : 0.0f;
    return __builtin_bit_cast(unsigned int, __builtin_amdgcn_cvt_pkrtz(a, b));
}

// Assemble a v16h A/B operand from two 16-byte chunks (global or LDS).
__device__ __forceinline__ v16h ldh16(const unsigned short* pa,
                                      const unsigned short* pb) {
    H16Bits t;
    t.a = *(const uint4*)pa;
    t.b = *(const uint4*)pb;
    return __builtin_bit_cast(v16h, t);
}

// Store 8 accumulator floats as relu->f16 packed 16B (one ds_store_b128)
__device__ __forceinline__ void store_relu_h16(char* dst, v8f d) {
    uint4 u;
    u.x = pack2_relu(d[0], d[1]);
    u.y = pack2_relu(d[2], d[3]);
    u.z = pack2_relu(d[4], d[5]);
    u.w = pack2_relu(d[6], d[7]);
    *(uint4*)dst = u;
}

// ---------------------------------------------------------------------------
// K1: dual 1x1 conv (128 -> 64) on 32x32 maps: f3p and asppp
// grid (4, 64, 8), block 256
// ---------------------------------------------------------------------------
__global__ __launch_bounds__(256) void k_conv1x1_dual(
    const float* __restrict__ aspp, const float* __restrict__ f3,
    const float* __restrict__ w_aspp, const float* __restrict__ w_f3,
    float* __restrict__ a_out, float* __restrict__ f_out) {
    const int pix = blockIdx.x * 256 + threadIdx.x;   // 0..1023
    const int o   = blockIdx.y;                        // 0..63
    const int n   = blockIdx.z;                        // 0..7
    const float* wa = w_aspp + o * 128;                // uniform per block
    const float* wf = w_f3   + o * 128;
    const float* ab = aspp + (size_t)n * 128 * 1024 + pix;
    const float* fb = f3   + (size_t)n * 128 * 1024 + pix;
    float sa = 0.0f, sf = 0.0f;
#pragma unroll 8
    for (int c = 0; c < 128; ++c) {
        sa += wa[c] * ab[(size_t)c * 1024];
        sf += wf[c] * fb[(size_t)c * 1024];
    }
    a_out[((size_t)n * 64 + o) * 1024 + pix] = sa;
    f_out[((size_t)n * 64 + o) * 1024 + pix] = sf;
}

// ---------------------------------------------------------------------------
// K2: fused bilinear-upsample(align_corners) + concat(132ch) + w_f9 matmul
//     + per-pixel L2 normalize + f16 store as F[n][pix][128]
// grid (4096, 8), block 128 (thread == output channel)
// ---------------------------------------------------------------------------
__global__ __launch_bounds__(128) void k_f9_norm(
    const float* __restrict__ inp, const float* __restrict__ a32,
    const float* __restrict__ f3b, const float* __restrict__ w9,
    unsigned short* __restrict__ F) {
    __shared__ float gv[132];
    __shared__ float red[128];
    const int pix = blockIdx.x;      // 0..4095
    const int n   = blockIdx.y;      // 0..7
    const int yy  = pix >> 6, xx = pix & 63;

    for (int c = threadIdx.x; c < 132; c += 128) {
        const float* src; int S;
        if (c < 4)       { src = inp + ((size_t)n * 4  + c)        * 262144; S = 512; }
        else if (c < 68) { src = a32 + ((size_t)n * 64 + (c - 4))  * 1024;   S = 32;  }
        else             { src = f3b + ((size_t)n * 64 + (c - 68)) * 1024;   S = 32;  }
        const float scale = (float)(S - 1) / 63.0f;
        const float fy = yy * scale, fx = xx * scale;
        int y0 = (int)fy, x0 = (int)fx;
        int y1 = y0 + 1; if (y1 > S - 1) y1 = S - 1;
        int x1 = x0 + 1; if (x1 > S - 1) x1 = S - 1;
        const float wy = fy - (float)y0, wx = fx - (float)x0;
        const float v0 = src[(size_t)y0 * S + x0] * (1.0f - wx) + src[(size_t)y0 * S + x1] * wx;
        const float v1 = src[(size_t)y1 * S + x0] * (1.0f - wx) + src[(size_t)y1 * S + x1] * wx;
        gv[c] = v0 * (1.0f - wy) + v1 * wy;
    }
    __syncthreads();

    const int o = threadIdx.x;       // output channel 0..127
    const float* w = w9 + (size_t)o * 132;
    float acc = 0.0f;
#pragma unroll 4
    for (int c = 0; c < 132; ++c) acc += w[c] * gv[c];

    red[o] = acc * acc;
    __syncthreads();
#pragma unroll
    for (int s = 64; s > 0; s >>= 1) {
        if (o < s) red[o] += red[o + s];
        __syncthreads();
    }
    const float nrm = sqrtf(red[0]);
    const float v = acc / (nrm + 1e-5f);
    F[(((size_t)n * 4096) + pix) * 128 + o] = f2h(v);
}

// ---------------------------------------------------------------------------
// K3: cam -> f16, padded to 32 rows: row 21 = ones (free column-sum),
//     rows 22..31 = 0. camb[n][32][4096]
// grid (16, 32, 8), block 256
// ---------------------------------------------------------------------------
__global__ __launch_bounds__(256) void k_prep_cam(
    const float* __restrict__ cam, unsigned short* __restrict__ camb) {
    const int p = blockIdx.x * 256 + threadIdx.x;  // 0..4095
    const int k = blockIdx.y;                       // 0..31
    const int n = blockIdx.z;                       // 0..7
    float v;
    if (k < 21)      v = cam[(((size_t)n * 21) + k) * 4096 + p];
    else if (k == 21) v = 1.0f;
    else              v = 0.0f;
    camb[(((size_t)n * 32) + k) * 4096 + p] = f2h(v);
}

// ---------------------------------------------------------------------------
// K4: fused affinity + column-normalize + cam propagation.
// Per wave: 32 q-columns (two 16-q strips sharing all A-side loads).
// Loop p in steps of 32:
//   S[32p x 32q] = Fp (A, f16) x Fq (B, f16), K=128 via 16x16x32 WMMA
//   relu(S) -> f16 -> LDS [q][p] (80B padded rows) -> reload as B operand
//   acc[32k x 32q] += camb (A) x S (B)   (row 21 of camb = ones -> colsum)
// Finally out[k][q] = acc[k][q] / (acc[21][q] + 1e-5), k < 21.
// Per p-step: 20 WMMAs vs 20 global b128 loads (A/cam tiles reused by both
// q-strips) -> balanced matrix/VMEM issue, everything L2-resident.
// grid (16, 8), block 256 (8 independent waves, no block barriers)
// ---------------------------------------------------------------------------
__global__ __launch_bounds__(256) void k_affinity_cam(
    const unsigned short* __restrict__ F, const unsigned short* __restrict__ camb,
    float* __restrict__ out) {
    __shared__ uint4 smemS[8 * 32 * 5];  // 8 waves * 32 q-rows * 80 bytes

    const int lane  = threadIdx.x & 31;
    const int wave  = threadIdx.x >> 5;
    const int lq    = lane & 15;      // q / row / col within tile
    const int lhalf = lane >> 4;      // which K/M half this lane carries
    const int n     = blockIdx.y;
    const int q0    = blockIdx.x * 256 + wave * 32;

    const unsigned short* Fn = F    + (size_t)n * 4096 * 128;
    const unsigned short* Cn = camb + (size_t)n * 32 * 4096;
    char* sS = (char*)(smemS + wave * 32 * 5);

    // Preload B operands for both q-strips: 4 chunks of K=32 channels each.
    v16h Bq0[4], Bq1[4];
    {
        const unsigned short* qa = Fn + (size_t)(q0 + lq) * 128 + lhalf * 8;
        const unsigned short* qb = qa + 16 * 128;
#pragma unroll
        for (int cc = 0; cc < 4; ++cc) {
            Bq0[cc] = ldh16(qa + cc * 32, qa + cc * 32 + 16);
            Bq1[cc] = ldh16(qb + cc * 32, qb + cc * 32 + 16);
        }
    }

    v8f acc0 = {}, acc1 = {};   // q-strip 0: k 0..15 / 16..31
    v8f acc2 = {}, acc3 = {};   // q-strip 1
    const unsigned short* Arow = Fn + (size_t)lq * 128 + lhalf * 8;   // + p*128
    const unsigned short* Crow = Cn + (size_t)lq * 4096 + lhalf * 8;  // + p

    for (int p0 = 0; p0 < 4096; p0 += 32) {
        v8f S0 = {}, S1 = {};   // strip 0: p 0..15 / 16..31
        v8f S2 = {}, S3 = {};   // strip 1
        const unsigned short* A0b = Arow + (size_t)p0 * 128;        // rows p0..p0+15
        const unsigned short* A1b = A0b + 16 * 128;                 // rows p0+16..+31
#pragma unroll
        for (int cc = 0; cc < 4; ++cc) {
            v16h A0 = ldh16(A0b + cc * 32, A0b + cc * 32 + 16);
            v16h A1 = ldh16(A1b + cc * 32, A1b + cc * 32 + 16);
            S0 = __builtin_amdgcn_wmma_f32_16x16x32_f16(false, A0, false, Bq0[cc],
                                                        (short)0, S0, false, false);
            S1 = __builtin_amdgcn_wmma_f32_16x16x32_f16(false, A1, false, Bq0[cc],
                                                        (short)0, S1, false, false);
            S2 = __builtin_amdgcn_wmma_f32_16x16x32_f16(false, A0, false, Bq1[cc],
                                                        (short)0, S2, false, false);
            S3 = __builtin_amdgcn_wmma_f32_16x16x32_f16(false, A1, false, Bq1[cc],
                                                        (short)0, S3, false, false);
        }

        // relu(S) -> f16 into LDS laid out [q][p] (p contiguous, 80B row pitch)
        char* row0 = sS + lq * 80;              // strip 0 rows
        char* row1 = sS + (16 + lq) * 80;       // strip 1 rows
        store_relu_h16(row0 + lhalf * 16,      S0);   // p = p0 + 0..15 half
        store_relu_h16(row0 + 32 + lhalf * 16, S1);   // p = p0 + 16..31 half
        store_relu_h16(row1 + lhalf * 16,      S2);
        store_relu_h16(row1 + 32 + lhalf * 16, S3);

        // Reload S as B operands (K = p within this 32-chunk, N = q)
        v16h BS0 = ldh16((const unsigned short*)(row0 + lhalf * 16),
                         (const unsigned short*)(row0 + lhalf * 16 + 32));
        v16h BS1 = ldh16((const unsigned short*)(row1 + lhalf * 16),
                         (const unsigned short*)(row1 + lhalf * 16 + 32));

        // cam A tiles: k rows 0..15 and 16..31, K = p chunk (shared by strips)
        v16h C0 = ldh16(Crow + p0,             Crow + p0 + 16);
        v16h C1 = ldh16(Crow + 16 * 4096 + p0, Crow + 16 * 4096 + p0 + 16);
        acc0 = __builtin_amdgcn_wmma_f32_16x16x32_f16(false, C0, false, BS0,
                                                      (short)0, acc0, false, false);
        acc1 = __builtin_amdgcn_wmma_f32_16x16x32_f16(false, C1, false, BS0,
                                                      (short)0, acc1, false, false);
        acc2 = __builtin_amdgcn_wmma_f32_16x16x32_f16(false, C0, false, BS1,
                                                      (short)0, acc2, false, false);
        acc3 = __builtin_amdgcn_wmma_f32_16x16x32_f16(false, C1, false, BS1,
                                                      (short)0, acc3, false, false);
    }

    // colsum[q] lives at k==21 -> acc1/acc3 vgpr 5, lanes 0..15 (lane == q)
    int cs0 = __builtin_amdgcn_ds_bpermute(lq << 2,
                                           __builtin_bit_cast(int, acc1[5]));
    int cs1 = __builtin_amdgcn_ds_bpermute(lq << 2,
                                           __builtin_bit_cast(int, acc3[5]));
    const float inv0 = 1.0f / (__builtin_bit_cast(float, cs0) + 1e-5f);
    const float inv1 = 1.0f / (__builtin_bit_cast(float, cs1) + 1e-5f);

    float* On0 = out + (size_t)n * 21 * 4096 + (q0 + lq);
    float* On1 = On0 + 16;
#pragma unroll
    for (int r = 0; r < 8; ++r) {
        const int k0 = r + 8 * lhalf;          // 0..15
        On0[(size_t)k0 * 4096] = acc0[r] * inv0;
        On1[(size_t)k0 * 4096] = acc2[r] * inv1;
        const int k1 = 16 + r + 8 * lhalf;     // 16..31, keep <21
        if (k1 < 21) {
            On0[(size_t)k1 * 4096] = acc1[r] * inv0;
            On1[(size_t)k1 * 4096] = acc3[r] * inv1;
        }
    }
}

// ---------------------------------------------------------------------------
// Launch
// ---------------------------------------------------------------------------
extern "C" void kernel_launch(void* const* d_in, const int* in_sizes, int n_in,
                              void* d_out, int out_size, void* d_ws, size_t ws_size,
                              hipStream_t stream) {
    (void)in_sizes; (void)n_in; (void)out_size; (void)ws_size;
    const float* aspp   = (const float*)d_in[0];  // [8,128,32,32]
    const float* f3     = (const float*)d_in[1];  // [8,128,32,32]
    const float* inp    = (const float*)d_in[2];  // [8,4,512,512]
    const float* cam    = (const float*)d_in[3];  // [8,21,64,64]
    const float* w_aspp = (const float*)d_in[4];  // [64,128]
    const float* w_f3   = (const float*)d_in[5];  // [64,128]
    const float* w_f9   = (const float*)d_in[6];  // [128,132]
    float* out = (float*)d_out;                   // [8,21,64,64]

    char* ws = (char*)d_ws;
    float*          a32  = (float*)ws;                              // 2 MB
    float*          f3b  = (float*)(ws + (size_t)(2u << 20));       // 2 MB
    unsigned short* F    = (unsigned short*)(ws + (size_t)(4u << 20));   // 8 MB
    unsigned short* camb = (unsigned short*)(ws + (size_t)(12u << 20));  // 2 MB

    k_conv1x1_dual<<<dim3(4, 64, 8), 256, 0, stream>>>(aspp, f3, w_aspp, w_f3,
                                                       a32, f3b);
    k_f9_norm<<<dim3(4096, 8), 128, 0, stream>>>(inp, a32, f3b, w_f9, F);
    k_prep_cam<<<dim3(16, 32, 8), 256, 0, stream>>>(cam, camb);
    k_affinity_cam<<<dim3(16, 8), 256, 0, stream>>>(F, camb, out);
}
